// CARAFE_downsampling_80547816669956
// MI455X (gfx1250) — compile-verified, matrix operand order checked by reading, not run on hardware
//
#include <hip/hip_runtime.h>
#include <hip/hip_bf16.h>

// CARAFE downsampling for MI455X (gfx1250, wave32, WMMA).
//
// Stages (all GEMM-shaped work routed through v_wmma_f32_16x16x32_f16):
//   k_down : 1x1 conv 64->64      GEMM M=131072 N=64 K=64   -> k1 (f16, NHWC)
//   k_enc  : 3x3 s2 conv 64->25   GEMM M=32768  N=32 K=576  + softmax -> k2 (f32)
//   k_reasm: CARAFE gather (VALU) -> t (f16, pixel-major [p][256])
//   k_out  : 1x1 conv 256->64     GEMM M=32768  N=64 K=256  -> out (f32, NCHW)

typedef __attribute__((ext_vector_type(16))) _Float16 v16h;
typedef __attribute__((ext_vector_type(8)))  float    v8f;

// Async global->LDS DMA path (ASYNCcnt), used when the toolchain exposes it.
#if defined(__has_builtin)
#if __has_builtin(__builtin_amdgcn_global_load_async_to_lds_b128) && \
    __has_builtin(__builtin_amdgcn_s_wait_asynccnt)
#define USE_ASYNC_LDS 1
#endif
#endif
#ifndef USE_ASYNC_LDS
#define USE_ASYNC_LDS 0
#endif

// Builtin signature (from compiler diagnostic): param0 = int4 addrspace(1)*
// (global source), param1 = int4 addrspace(3)* (LDS dest), then imm offset,
// imm cpol.
typedef int v4i __attribute__((vector_size(16)));
typedef __attribute__((address_space(1))) v4i gv4i;
typedef __attribute__((address_space(3))) v4i lv4i;

union FragU { v16h h; unsigned u[8]; };

// Load one lane's share of a 16x32 f16 A/B fragment (CDNA5 ISA 7.12.2):
// lanes 0-15: VGPR0-3 hold K=0..7, VGPR4-7 hold K=16..23;
// lanes 16-31: K=8..15 and K=24..31.  rowBase points at this lane's
// (M or N) row with 32 contiguous K halves (4-byte aligned).
__device__ __forceinline__ v16h frag_load(const _Float16* rowBase) {
  const int g = (threadIdx.x >> 4) & 1;
  const unsigned* p = (const unsigned*)(rowBase + 8 * g);
  FragU r;
#pragma unroll
  for (int i = 0; i < 4; ++i) r.u[i] = p[i];
#pragma unroll
  for (int i = 0; i < 4; ++i) r.u[4 + i] = p[8 + i];
  return r.h;
}

#define HW 65536   // 256*256
#define HWD 16384  // 128*128

// ---------------- Stage 1: 1x1 down conv (WMMA) ----------------
__global__ __launch_bounds__(256) void k_down(const float* __restrict__ x,
                                              const float* __restrict__ dw,
                                              const float* __restrict__ db,
                                              _Float16* __restrict__ k1) {
  __shared__ _Float16 Xs[128 * 64];  // 128 pixels x 64 in-ch
  __shared__ _Float16 Bs[64 * 64];   // Bt[cm][c] = dw[cm][c]
  const int tid = threadIdx.x;
  const int bx  = blockIdx.x;
  const int n   = bx >> 9;            // 512 blocks per image
  const int p0  = (bx & 511) * 128;

  __builtin_prefetch(x + (size_t)n * 64 * HW + p0, 0, 1);

  for (int idx = tid; idx < 128 * 64; idx += 256) {
    int c = idx >> 7, p = idx & 127;
    Xs[p * 64 + c] = (_Float16)x[((size_t)(n * 64 + c)) * HW + p0 + p];
  }
  for (int idx = tid; idx < 64 * 64; idx += 256)
    Bs[idx] = (_Float16)dw[idx];
  __syncthreads();

  const int lane = tid & 31;
  const int wave = tid >> 5;
  const int m0   = wave * 16;
  const int mrow = lane & 15;

  v16h a0 = frag_load(&Xs[(m0 + mrow) * 64 + 0]);
  v16h a1 = frag_load(&Xs[(m0 + mrow) * 64 + 32]);

  for (int nt = 0; nt < 4; ++nt) {
    const int n0 = nt * 16;
    v16h b0 = frag_load(&Bs[(n0 + mrow) * 64 + 0]);
    v16h b1 = frag_load(&Bs[(n0 + mrow) * 64 + 32]);
    v8f acc = {};
    acc = __builtin_amdgcn_wmma_f32_16x16x32_f16(false, a0, false, b0, (short)0, acc, false, false);
    acc = __builtin_amdgcn_wmma_f32_16x16x32_f16(false, a1, false, b1, (short)0, acc, false, false);
    const int   cm   = n0 + (lane & 15);
    const float bias = db[cm];
#pragma unroll
    for (int v = 0; v < 8; ++v) {
      int mm = v + ((lane >> 4) << 3);
      k1[((size_t)(n * HW + p0 + m0 + mm)) * 64 + cm] = (_Float16)(acc[v] + bias);
    }
  }
}

// ---------------- Stage 2: 3x3 s2 enc conv + softmax (WMMA) ----------------
// Tile: 4x8 output pixels (32) per block, 128 threads (4 waves).
// inT staging is a pure f16 copy -> async global->LDS DMA when available.
__global__ __launch_bounds__(128) void k_enc(const _Float16* __restrict__ k1,
                                             const float* __restrict__ ew,
                                             const float* __restrict__ eb,
                                             float* __restrict__ k2) {
  __shared__ __align__(16) _Float16 inT[9 * 17 * 64];  // halo'd input tile (NHWC)
  __shared__ _Float16 Bt[32 * 576];      // Bt[j][tap*64+c], rows 25..31 = 0
  __shared__ float    eSm[32 * 32];
  const int tid = threadIdx.x;
  const int bx  = blockIdx.x;
  const int n   = bx >> 9;
  const int r   = bx & 511;
  const int hb  = r >> 4;   // hd tile base / 4
  const int wb  = r & 15;   // wd tile base / 8
  const int gh0 = hb * 8 - 1;
  const int gw0 = wb * 16 - 1;

  // Stage halo'd 9x17x64 f16 tile in 16-byte chunks (8 halves per chunk).
  for (int idx = tid; idx < 9 * 17 * 8; idx += 128) {
    int pix = idx >> 3, vv = idx & 7;
    int pr = pix / 17, pc = pix % 17;
    int gh = gh0 + pr, gw = gw0 + pc;
    _Float16* ldst = &inT[pix * 64 + vv * 8];
    if (gh >= 0 && gh < 256 && gw >= 0 && gw < 256) {
      const _Float16* gsrc = &k1[((size_t)(n * 256 + gh) * 256 + gw) * 64 + vv * 8];
#if USE_ASYNC_LDS
      __builtin_amdgcn_global_load_async_to_lds_b128((gv4i*)gsrc, (lv4i*)ldst, 0, 0);
#else
      *(uint4*)ldst = *(const uint4*)gsrc;
#endif
    } else {
      uint4 z = {0u, 0u, 0u, 0u};
      *(uint4*)ldst = z;
    }
  }
  for (int idx = tid; idx < 32 * 576; idx += 128) {
    int j = idx / 576, k = idx % 576;
    int tap = k >> 6, c = k & 63;
    Bt[idx] = (j < 25) ? (_Float16)ew[(j * 64 + c) * 9 + tap] : (_Float16)0.f;
  }
#if USE_ASYNC_LDS
  __builtin_amdgcn_s_wait_asynccnt(0);
#endif
  __syncthreads();

  const int lane = tid & 31;
  const int wave = tid >> 5;
  const int m0   = (wave >> 1) * 16;
  const int n0   = (wave & 1) * 16;
  const int mrow = lane & 15;
  const int pm   = m0 + mrow;       // pixel 0..31 in tile
  const int rr   = pm >> 3, cc = pm & 7;

  v8f acc = {};
#pragma unroll
  for (int ks = 0; ks < 18; ++ks) {
    const int tap = ks >> 1, cbase = (ks & 1) * 32;
    const int dy = tap / 3, dx = tap % 3;
    v16h a = frag_load(&inT[((2 * rr + dy) * 17 + (2 * cc + dx)) * 64 + cbase]);
    v16h b = frag_load(&Bt[(n0 + mrow) * 576 + tap * 64 + cbase]);
    acc = __builtin_amdgcn_wmma_f32_16x16x32_f16(false, a, false, b, (short)0, acc, false, false);
  }
#pragma unroll
  for (int v = 0; v < 8; ++v) {
    int mm = v + ((lane >> 4) << 3);
    eSm[(m0 + mm) * 32 + n0 + (lane & 15)] = acc[v];
  }
  __syncthreads();

  if (tid < 32) {
    const int hd = hb * 4 + (tid >> 3);
    const int wd = wb * 8 + (tid & 7);
    float e[25], mx = -1e30f;
#pragma unroll
    for (int j = 0; j < 25; ++j) { e[j] = eSm[tid * 32 + j] + eb[j]; mx = fmaxf(mx, e[j]); }
    float s = 0.f;
#pragma unroll
    for (int j = 0; j < 25; ++j) { e[j] = __expf(e[j] - mx); s += e[j]; }
    const float inv = 1.f / s;
    float* dst = &k2[((size_t)n * HWD + hd * 128 + wd) * 32];
#pragma unroll
    for (int j = 0; j < 25; ++j) dst[j] = e[j] * inv;
  }
}

// ---------------- Stage 3: CARAFE weighted reassembly (VALU gather) ----------------
// Faithful row-major reshape mapping: c = c2>>2, h = (c2&3)*64 + hd/2,
// w = (hd&1)*128 + wd; 5x5 window, zero pad 2.
__global__ __launch_bounds__(256) void k_reasm(const float* __restrict__ x,
                                               const float* __restrict__ k2,
                                               _Float16* __restrict__ t) {
  __shared__ float kv[25];
  const int bx = blockIdx.x;
  const int n  = bx >> 14;
  const int q  = bx & 16383;
  const int hd = q >> 7, wd = q & 127;
  const int tid = threadIdx.x;
  if (tid < 25) kv[tid] = k2[((size_t)n * HWD + q) * 32 + tid];
  __syncthreads();

  const int c2 = tid;
  const int c  = c2 >> 2, qd = c2 & 3;
  const int hbase = qd * 64 + (hd >> 1) - 2;
  const int wbase = (hd & 1) * 128 + wd - 2;
  const float* xc = &x[((size_t)(n * 64 + c)) * HW];
  float acc = 0.f;
#pragma unroll
  for (int i = 0; i < 5; ++i) {
    int h = hbase + i;
    if (h < 0 || h >= 256) continue;
#pragma unroll
    for (int j = 0; j < 5; ++j) {
      int w = wbase + j;
      if (w < 0 || w >= 256) continue;
      acc += xc[h * 256 + w] * kv[i * 5 + j];
    }
  }
  t[((size_t)n * HWD + q) * 256 + c2] = (_Float16)acc;
}

// ---------------- Stage 4: 1x1 out conv (WMMA, A streamed from global) ----------------
__global__ __launch_bounds__(256) void k_out(const _Float16* __restrict__ t,
                                             const float* __restrict__ ow,
                                             const float* __restrict__ ob,
                                             float* __restrict__ out) {
  __shared__ _Float16 Bt[64 * 256];  // Bt[oc][c2] = ow[oc][c2]
  const int tid = threadIdx.x;
  const int bx  = blockIdx.x;
  const int n   = bx >> 7;
  const int p0  = (bx & 127) * 128;

  for (int idx = tid; idx < 64 * 256; idx += 256)
    Bt[idx] = (_Float16)ow[idx];
  __syncthreads();

  const int lane = tid & 31;
  const int wave = tid >> 5;
  const int m0   = wave * 16;
  const int mrow = lane & 15;

  // t is pixel-major with 256 contiguous K halves -> A fragments direct from global.
  v16h aF[8];
  const _Float16* arow = &t[((size_t)n * HWD + p0 + m0 + mrow) * 256];
#pragma unroll
  for (int ks = 0; ks < 8; ++ks) aF[ks] = frag_load(arow + ks * 32);

  for (int nt = 0; nt < 4; ++nt) {
    const int n0 = nt * 16;
    v8f acc = {};
#pragma unroll
    for (int ks = 0; ks < 8; ++ks) {
      v16h b = frag_load(&Bt[(n0 + mrow) * 256 + ks * 32]);
      acc = __builtin_amdgcn_wmma_f32_16x16x32_f16(false, aF[ks], false, b, (short)0, acc, false, false);
    }
    const int   oc   = n0 + (lane & 15);
    const float bias = ob[oc];
#pragma unroll
    for (int v = 0; v < 8; ++v) {
      int mm = v + ((lane >> 4) << 3);
      out[((size_t)(n * 64 + oc)) * HWD + p0 + m0 + mm] = acc[v] + bias;
    }
  }
}

extern "C" void kernel_launch(void* const* d_in, const int* in_sizes, int n_in,
                              void* d_out, int out_size, void* d_ws, size_t ws_size,
                              hipStream_t stream) {
  (void)in_sizes; (void)n_in; (void)out_size; (void)ws_size;
  const float* x  = (const float*)d_in[0];
  const float* dw = (const float*)d_in[1];
  const float* db = (const float*)d_in[2];
  const float* ew = (const float*)d_in[3];
  const float* eb = (const float*)d_in[4];
  const float* ow = (const float*)d_in[5];
  const float* ob = (const float*)d_in[6];
  float* out = (float*)d_out;

  // Workspace partition (bytes): k1 f16 16 MiB | k2 f32 4 MiB | t f16 16 MiB
  char* ws = (char*)d_ws;
  _Float16* k1 = (_Float16*)ws;
  float*    k2 = (float*)(ws + (size_t)16777216);
  _Float16* tt = (_Float16*)(ws + (size_t)20971520);

  k_down <<<dim3(1024),  dim3(256), 0, stream>>>(x, dw, db, k1);
  k_enc  <<<dim3(1024),  dim3(128), 0, stream>>>(k1, ew, eb, k2);
  k_reasm<<<dim3(32768), dim3(256), 0, stream>>>(x, k2, tt);
  k_out  <<<dim3(256),   dim3(256), 0, stream>>>(tt, ow, ob, out);
}
